// DeepAugmentedMUSIC_4466765987883
// MI455X (gfx1250) — compile-verified, matrix operand order checked by reading, not run on hardware
//
#include <hip/hip_runtime.h>
#include <hip/hip_bf16.h>
#include <stdint.h>

typedef __attribute__((ext_vector_type(16))) __bf16 v16bf;
typedef __attribute__((ext_vector_type(8)))  float  v8f;
typedef int v4i __attribute__((vector_size(16)));

union BF16x16 { v16bf v; uint32_t d[8]; unsigned short u[16]; };

#if defined(__HIP_DEVICE_COMPILE__) && __has_builtin(__builtin_amdgcn_global_load_async_to_lds_b128)
#define ASYNC_LDS 1
#endif

__device__ inline void async_b128(const void* gsrc, void* ldst) {
#if defined(ASYNC_LDS)
    __builtin_amdgcn_global_load_async_to_lds_b128(
        (__attribute__((address_space(1))) v4i*)(v4i*)(void*)gsrc,
        (__attribute__((address_space(3))) v4i*)(v4i*)ldst, 0, 0);
#else
    (void)gsrc; (void)ldst;
#endif
}

__device__ inline void wait_async0() {
#if defined(ASYNC_LDS)
#if __has_builtin(__builtin_amdgcn_s_wait_asynccnt)
    __builtin_amdgcn_s_wait_asynccnt(0);
#else
    asm volatile("s_wait_asynccnt 0x0" ::: "memory");
#endif
#endif
}

__device__ inline unsigned short f2bfbits(float f) {
    union { float f; uint32_t u; } x; x.f = f;
    uint32_t u = x.u;
    uint32_t r = u + 0x7FFFu + ((u >> 16) & 1u);
    return (unsigned short)(r >> 16);
}

// Fast transcendental forms: single v_exp_f32 + v_rcp_f32, no IEEE divide chain.
__device__ inline float fast_sigmoid(float x) {
    return __builtin_amdgcn_rcpf(1.f + __expf(-x));
}
__device__ inline float fast_tanh(float x) {
    return 1.f - 2.f * __builtin_amdgcn_rcpf(1.f + __expf(2.f * x));
}

__device__ inline v8f wmma_bf16(v16bf a, v16bf b, v8f c) {
    // D = A(16x32 bf16) * B(32x16 bf16) + C(16x16 f32)
    return __builtin_amdgcn_wmma_f32_16x16x32_bf16(
        false, a, false, b, (short)0, c, false, false);
}

// ---------------------------------------------------------------------------
// Repack weight W (Nr x K, f32 row-major) into WMMA B-fragment order, bf16:
// out[nt][kt][lane][e] = bf16(W[nt*16 + lane%16, kt*32 + e + 8*(lane/16) + (e>=8?8:0)])
// ---------------------------------------------------------------------------
__global__ void repack_w_kernel(const float* __restrict__ W, int Nr, int K,
                                int nTiles, int kTiles, unsigned short* __restrict__ out) {
    int i = blockIdx.x * 256 + threadIdx.x;
    int tot = nTiles * kTiles * 512;
    if (i >= tot) return;
    int e    = i & 15;
    int lane = (i >> 4) & 31;
    int kt   = (i >> 9) % kTiles;
    int nt   = i / (512 * kTiles);
    int n = nt * 16 + (lane & 15);
    int k = kt * 32 + (e < 8 ? e : e + 8) + 8 * (lane >> 4);
    unsigned short v = 0;
    if (n < Nr && k < K) v = f2bfbits(W[(size_t)n * K + k]);
    out[i] = v;
}

// ---------------------------------------------------------------------------
// BatchNorm statistics per t.
// ---------------------------------------------------------------------------
__global__ __launch_bounds__(256) void bn_stats_kernel(
    const float* __restrict__ Xr, const float* __restrict__ Xi,
    const float* __restrict__ gamma, const float* __restrict__ beta,
    float* __restrict__ scale, float* __restrict__ shift) {
    int t = blockIdx.x;
    int tid = threadIdx.x;
    const float* base = (t < 512) ? (Xr + (size_t)t * 128)
                                  : (Xi + (size_t)(t - 512) * 128);
    float s = 0.f, s2 = 0.f;
    for (int i = tid; i < 256 * 128; i += 256) {
        int b = i >> 7, h = i & 127;
        float v = base[(size_t)b * 65536 + h];
        s += v; s2 += v * v;
    }
    __shared__ float rs[256], rs2[256];
    rs[tid] = s; rs2[tid] = s2;
    __syncthreads();
    for (int off = 128; off; off >>= 1) {
        if (tid < off) { rs[tid] += rs[tid + off]; rs2[tid] += rs2[tid + off]; }
        __syncthreads();
    }
    if (tid == 0) {
        float mean = rs[0] * (1.f / 32768.f);
        float var  = rs2[0] * (1.f / 32768.f) - mean * mean;
        float inv  = rsqrtf(var + 1e-5f);
        float sc   = inv * gamma[t];
        scale[t] = sc;
        shift[t] = beta[t] - mean * sc;
    }
}

// Normalize + convert to bf16, layout Xn[b][t][h].
__global__ void bn_norm_kernel(const float* __restrict__ Xr, const float* __restrict__ Xi,
                               const float* __restrict__ scale, const float* __restrict__ shift,
                               unsigned short* __restrict__ Xn) {
    size_t idx = (size_t)blockIdx.x * 256 + threadIdx.x;
    if (idx >= (size_t)256 * 1024 * 128) return;
    int h = (int)(idx & 127);
    size_t r = idx >> 7;
    int t = (int)(r % 1024);
    size_t b = r / 1024;
    size_t i = (size_t)t * 128 + h;
    float v = (i < 65536) ? Xr[b * 65536 + i] : Xi[b * 65536 + i - 65536];
    v = v * scale[t] + shift[t];
    Xn[idx] = f2bfbits(v);
}

// ---------------------------------------------------------------------------
// Fused GRU. One block per 16 batch rows; weights live in VGPRs (192 VGPRs of
// B-fragments per wave, loaded once); x tiles double-buffered into LDS via
// async global->LDS one step ahead of the recurrent WMMA chain.
// ---------------------------------------------------------------------------
__global__ __launch_bounds__(256, 1) void gru_kernel(
    const unsigned short* __restrict__ Xn,        // (256,1024,128) bf16
    const unsigned short* __restrict__ wih_pk,
    const unsigned short* __restrict__ whh_pk,
    const float* __restrict__ b_ih, const float* __restrict__ b_hh,
    unsigned short* __restrict__ hT_out)          // (256,128) bf16
{
    __shared__ float g[16 * 256];
    __shared__ float gxn[16 * 128];
    __shared__ float ghn[16 * 128];
    __shared__ float hf[16 * 128];
    __shared__ unsigned short hb[16 * 128];
#if defined(ASYNC_LDS)
    __shared__ unsigned short xstage[2][16 * 128];
#endif

    int tid = threadIdx.x;
    int wave = tid >> 5, lane = tid & 31;
    int col16 = lane & 15, hl = lane >> 4;
    int r0 = blockIdx.x * 16;

    // ---- hoist all weight B-fragments into registers (loop-invariant) ----
    const uint32_t* wihq = (const uint32_t*)wih_pk;
    const uint32_t* whhq = (const uint32_t*)whh_pk;
    BF16x16 wih[3][4], whh[3][4];
#pragma unroll
    for (int it = 0; it < 3; ++it) {
        int tl = wave * 3 + it;
#pragma unroll
        for (int kt = 0; kt < 4; ++kt) {
            const uint32_t* p1 = wihq + ((size_t)(tl * 4 + kt) * 32 + lane) * 8;
            const uint32_t* p2 = whhq + ((size_t)(tl * 4 + kt) * 32 + lane) * 8;
#pragma unroll
            for (int i = 0; i < 8; ++i) { wih[it][kt].d[i] = p1[i]; whh[it][kt].d[i] = p2[i]; }
        }
    }
    float bsum[3], bxv[3], bhv[3];
#pragma unroll
    for (int it = 0; it < 3; ++it) {
        int col = (wave * 3 + it) * 16 + col16;
        float a = b_ih[col], b = b_hh[col];
        bsum[it] = a + b; bxv[it] = a; bhv[it] = b;
    }

    for (int i = tid; i < 16 * 128; i += 256) { hf[i] = 0.f; hb[i] = 0; }

#if defined(ASYNC_LDS)
    {   // stage x tile for t=0
        int row = tid >> 4, ch = tid & 15;
        const unsigned short* src = Xn + ((size_t)(r0 + row) * 1024) * 128 + ch * 8;
        async_b128(src, &xstage[0][row * 128 + ch * 8]);
        wait_async0();
    }
#endif
    __syncthreads();

    const unsigned short* xrow = Xn + ((size_t)(r0 + col16) * 1024) * 128;
    (void)xrow;

#pragma unroll 1
    for (int t = 0; t < 1024; ++t) {
#if defined(ASYNC_LDS)
        if (t + 1 < 1024) {   // prefetch next step's x tile (overlapped)
            int row = tid >> 4, ch = tid & 15;
            const unsigned short* src =
                Xn + ((size_t)(r0 + row) * 1024 + (t + 1)) * 128 + ch * 8;
            async_b128(src, &xstage[(t + 1) & 1][row * 128 + ch * 8]);
        }
#endif
        // A-fragments: M = lane%16 (local batch row), K pattern per ISA 7.12.2
        BF16x16 ax[4], ah[4];
#if !defined(ASYNC_LDS)
        const unsigned short* xr = xrow + (size_t)t * 128;
#endif
#pragma unroll
        for (int kt = 0; kt < 4; ++kt) {
#pragma unroll
            for (int i = 0; i < 8; ++i) {
                int koff = kt * 32 + (i < 4 ? 2 * i : 2 * i + 8) + 8 * hl;
#if defined(ASYNC_LDS)
                ax[kt].d[i] = *(const uint32_t*)(&xstage[t & 1][col16 * 128 + koff]);
#else
                ax[kt].d[i] = *(const uint32_t*)(xr + koff);
#endif
                ah[kt].d[i] = *(const uint32_t*)(&hb[col16 * 128 + koff]);
            }
        }
#pragma unroll
        for (int it = 0; it < 3; ++it) {
            int tl = wave * 3 + it;        // uniform per wave
            int col = tl * 16 + col16;
            if (tl < 16) {                 // r and z gates: single sum
                v8f acc;
#pragma unroll
                for (int r = 0; r < 8; ++r) acc[r] = bsum[it];
#pragma unroll
                for (int kt = 0; kt < 4; ++kt) acc = wmma_bf16(ax[kt].v, wih[it][kt].v, acc);
#pragma unroll
                for (int kt = 0; kt < 4; ++kt) acc = wmma_bf16(ah[kt].v, whh[it][kt].v, acc);
#pragma unroll
                for (int r = 0; r < 8; ++r)
                    g[(r + 8 * hl) * 256 + col] = acc[r];
            } else {                       // n gate: keep x/h parts separate
                v8f accx, acch;
#pragma unroll
                for (int r = 0; r < 8; ++r) { accx[r] = bxv[it]; acch[r] = bhv[it]; }
#pragma unroll
                for (int kt = 0; kt < 4; ++kt) accx = wmma_bf16(ax[kt].v, wih[it][kt].v, accx);
#pragma unroll
                for (int kt = 0; kt < 4; ++kt) acch = wmma_bf16(ah[kt].v, whh[it][kt].v, acch);
                int c = col - 256;
#pragma unroll
                for (int r = 0; r < 8; ++r) {
                    gxn[(r + 8 * hl) * 128 + c] = accx[r];
                    ghn[(r + 8 * hl) * 128 + c] = acch[r];
                }
            }
        }
        __syncthreads();
        // gate update: h' = (1-z)*n + z*h   (hardware rcp/exp forms)
        for (int i = tid; i < 16 * 128; i += 256) {
            int row = i >> 7, c = i & 127;
            float rr = fast_sigmoid(g[row * 256 + c]);
            float zz = fast_sigmoid(g[row * 256 + 128 + c]);
            float nn = fast_tanh(gxn[i] + rr * ghn[i]);
            float hnew = (1.f - zz) * nn + zz * hf[i];
            hf[i] = hnew;
            hb[i] = f2bfbits(hnew);
        }
        wait_async0();     // next x tile staged before anyone reads it
        __syncthreads();
    }
    for (int i = tid; i < 16 * 128; i += 256) {
        int row = i >> 7, c = i & 127;
        hT_out[(size_t)(r0 + row) * 128 + c] = hb[i];
    }
}

// ---------------------------------------------------------------------------
// Generic bf16 WMMA GEMM: out(256 x nTiles*16) = A(256 x kTiles*32) @ Wpk^T + bias.
// ---------------------------------------------------------------------------
__global__ __launch_bounds__(256) void gemm_bf16_kernel(
    const unsigned short* __restrict__ A, int lda,
    const unsigned short* __restrict__ Wpk,
    const float* __restrict__ bias,
    int nTiles, int kTiles,
    float* outF, int strideF,
    unsigned short* outB, int strideB,
    int nValid, int relu)
{
    int tid = threadIdx.x;
    int wid = (blockIdx.x * 256 + tid) >> 5;
    int lane = tid & 31;
    int total = 16 * nTiles;
    if (wid >= total) return;              // wave-uniform
    int nt = wid % nTiles, rt = wid / nTiles;
    int col16 = lane & 15, hl = lane >> 4;
    int arow = rt * 16 + col16;
    int col = nt * 16 + col16;

    v8f acc;
    float bv = (col < nValid) ? bias[col] : 0.f;
#pragma unroll
    for (int r = 0; r < 8; ++r) acc[r] = bv;

    const uint32_t* wq = (const uint32_t*)Wpk;
    for (int kt = 0; kt < kTiles; ++kt) {
        BF16x16 a, w;
        const unsigned short* ap = A + (size_t)arow * lda + kt * 32;
#pragma unroll
        for (int i = 0; i < 8; ++i)
            a.d[i] = *(const uint32_t*)(ap + (i < 4 ? 2 * i : 2 * i + 8) + 8 * hl);
        const uint32_t* wp = wq + ((size_t)(nt * kTiles + kt) * 32 + lane) * 8;
#pragma unroll
        for (int i = 0; i < 8; ++i) w.d[i] = wp[i];
        acc = wmma_bf16(a.v, w.v, acc);
    }
#pragma unroll
    for (int r = 0; r < 8; ++r) {
        int orow = rt * 16 + r + 8 * hl;
        float v = acc[r];
        if (relu) v = fmaxf(v, 0.f);
        if (col < nValid) {
            if (outF) outF[(size_t)orow * strideF + col] = v;
            if (outB) outB[(size_t)orow * strideB + col] = f2bfbits(v);
        }
    }
}

// ---------------------------------------------------------------------------
// Dominant-M subspace per batch via orthogonalized power iteration on the
// complex 64x64 K. MUSIC noise energy = ||s||^2 - ||Us^H s||^2.
// K loaded into LDS with async global->LDS b128.
// ---------------------------------------------------------------------------
__global__ __launch_bounds__(256) void subspace_kernel(
    const float* __restrict__ Rx, float* __restrict__ Us) {
    __shared__ float Kr[64 * 64], Ki[64 * 64];
    __shared__ float Vr[64 * 8], Vi[64 * 8];
    __shared__ float Wr[64 * 8], Wi[64 * 8];
    int b = blockIdx.x, tid = threadIdx.x;
    int wave = tid >> 5, lane = tid & 31;
    const float* R = Rx + (size_t)b * 8192;
#if defined(ASYNC_LDS)
#pragma unroll
    for (int c = 0; c < 4; ++c) {
        int j = (tid + c * 256) * 4;           // 16B chunks
        async_b128(R + j, &Kr[j]);
        async_b128(R + 4096 + j, &Ki[j]);
    }
#else
    for (int i = tid; i < 4096; i += 256) { Kr[i] = R[i]; Ki[i] = R[4096 + i]; }
#endif
    for (int i = tid; i < 512; i += 256) {
        int n = i >> 3, k = i & 7;
        float ph = 0.37f * n + 1.31f * k + 0.5f;
        Vr[i] = __cosf(ph); Vi[i] = __sinf(ph);
    }
    wait_async0();
    __syncthreads();
#pragma unroll 1
    for (int it = 0; it < 10; ++it) {
        for (int i = tid; i < 512; i += 256) {
            int n = i >> 3, k = i & 7;
            float ar = 0.f, ai = 0.f;
            for (int m = 0; m < 64; ++m) {
                float kr = Kr[n * 64 + m], ki = Ki[n * 64 + m];
                float vr = Vr[m * 8 + k], vi = Vi[m * 8 + k];
                ar += kr * vr - ki * vi;
                ai += kr * vi + ki * vr;
            }
            Wr[i] = ar; Wi[i] = ai;
        }
        __syncthreads();
        if (wave == 0) {                  // modified Gram-Schmidt, wave 0
            for (int k = 0; k < 8; ++k) {
                for (int j = 0; j < k; ++j) {
                    float dr = 0.f, di = 0.f;
                    for (int s = 0; s < 2; ++s) {
                        int n = lane + 32 * s;
                        float vr = Vr[n * 8 + j], vi = Vi[n * 8 + j];
                        float wr = Wr[n * 8 + k], wi = Wi[n * 8 + k];
                        dr += vr * wr + vi * wi;
                        di += vr * wi - vi * wr;
                    }
                    for (int off = 16; off; off >>= 1) {
                        dr += __shfl_down(dr, off, 32);
                        di += __shfl_down(di, off, 32);
                    }
                    dr = __shfl(dr, 0, 32); di = __shfl(di, 0, 32);
                    for (int s = 0; s < 2; ++s) {
                        int n = lane + 32 * s;
                        float vr = Vr[n * 8 + j], vi = Vi[n * 8 + j];
                        Wr[n * 8 + k] -= dr * vr - di * vi;
                        Wi[n * 8 + k] -= dr * vi + di * vr;
                    }
                }
                float nr = 0.f;
                for (int s = 0; s < 2; ++s) {
                    int n = lane + 32 * s;
                    nr += Wr[n * 8 + k] * Wr[n * 8 + k] + Wi[n * 8 + k] * Wi[n * 8 + k];
                }
                for (int off = 16; off; off >>= 1) nr += __shfl_down(nr, off, 32);
                nr = __shfl(nr, 0, 32);
                float inv = rsqrtf(nr + 1e-20f);
                for (int s = 0; s < 2; ++s) {
                    int n = lane + 32 * s;
                    Vr[n * 8 + k] = Wr[n * 8 + k] * inv;
                    Vi[n * 8 + k] = Wi[n * 8 + k] * inv;
                }
            }
        }
        __syncthreads();
    }
    for (int i = tid; i < 512; i += 256) {
        Us[((size_t)b * 512 + i) * 2]     = Vr[i];
        Us[((size_t)b * 512 + i) * 2 + 1] = Vi[i];
    }
}

// Steering-vector table sv[a][n] = exp(-i*pi*n*sin(ang_a)).
__global__ void sv_kernel(float* __restrict__ svr, float* __restrict__ svi) {
    int i = blockIdx.x * 256 + threadIdx.x;
    if (i >= 361 * 64) return;
    int a = i / 64, n = i % 64;
    float ang = -1.5707963267948966f + (float)a * (3.14159265358979f / 360.f);
    float ph = -3.14159265358979f * (float)n * __sinf(ang);
    svr[i] = __cosf(ph);
    svi[i] = __sinf(ph);
}

// spectrum[b,a] = 1 / (||s_a||^2 - ||Us^H s_a||^2); emit bf16, K-padded to 384.
__global__ void spectrum_kernel(const float* __restrict__ Us,
                                const float* __restrict__ svr,
                                const float* __restrict__ svi,
                                unsigned short* __restrict__ specb) {
    int b = blockIdx.y;
    int a = blockIdx.x * 128 + threadIdx.x;   // 0..383
    if (a >= 384) return;
    if (a >= 361) { specb[(size_t)b * 384 + a] = 0; return; }
    const float* U = Us + (size_t)b * 1024;
    float eq = 64.f;
    for (int k = 0; k < 8; ++k) {
        float cr = 0.f, ci = 0.f;
        for (int n = 0; n < 64; ++n) {
            float ur = U[(n * 8 + k) * 2], ui = U[(n * 8 + k) * 2 + 1];
            float sr = svr[a * 64 + n],   si = svi[a * 64 + n];
            cr += ur * sr + ui * si;
            ci += ur * si - ui * sr;
        }
        eq -= cr * cr + ci * ci;
    }
    eq = fmaxf(eq, 1e-6f);
    specb[(size_t)b * 384 + a] = f2bfbits(__builtin_amdgcn_rcpf(eq));
}

// ---------------------------------------------------------------------------
extern "C" void kernel_launch(void* const* d_in, const int* in_sizes, int n_in,
                              void* d_out, int out_size, void* d_ws, size_t ws_size,
                              hipStream_t stream) {
    const float* Xr    = (const float*)d_in[0];
    const float* Xi    = (const float*)d_in[1];
    const float* gamma = (const float*)d_in[2];
    const float* beta  = (const float*)d_in[3];
    const float* w_ih  = (const float*)d_in[4];
    const float* w_hh  = (const float*)d_in[5];
    const float* b_ih  = (const float*)d_in[6];
    const float* b_hh  = (const float*)d_in[7];
    const float* fc_w  = (const float*)d_in[8];
    const float* fc_b  = (const float*)d_in[9];
    const float* fc1_w = (const float*)d_in[10];
    const float* fc1_b = (const float*)d_in[11];
    const float* fc2_w = (const float*)d_in[12];
    const float* fc2_b = (const float*)d_in[13];
    const float* fc3_w = (const float*)d_in[14];
    const float* fc3_b = (const float*)d_in[15];

    char* ws = (char*)d_ws;
    size_t off = 0;
    auto alloc = [&](size_t bytes) -> void* {
        void* p = ws + off;
        off += (bytes + 255) & ~(size_t)255;
        return p;
    };

    unsigned short* wih_pk = (unsigned short*)alloc(24 * 4 * 512 * 2);
    unsigned short* whh_pk = (unsigned short*)alloc(24 * 4 * 512 * 2);
    unsigned short* fcw_pk = (unsigned short*)alloc((size_t)512 * 4 * 512 * 2);
    unsigned short* f1w_pk = (unsigned short*)alloc(8 * 12 * 512 * 2);
    unsigned short* f2w_pk = (unsigned short*)alloc(8 * 4 * 512 * 2);
    unsigned short* f3w_pk = (unsigned short*)alloc(1 * 4 * 512 * 2);
    float* scale = (float*)alloc(1024 * 4);
    float* shift = (float*)alloc(1024 * 4);
    unsigned short* Xn = (unsigned short*)alloc((size_t)256 * 1024 * 128 * 2);
    unsigned short* hT = (unsigned short*)alloc(256 * 128 * 2);
    float* Rx = (float*)alloc((size_t)256 * 8192 * 4);
    float* Us = (float*)alloc((size_t)256 * 512 * 2 * 4);
    float* svr = (float*)alloc(361 * 64 * 4);
    float* svi = (float*)alloc(361 * 64 * 4);
    unsigned short* spec = (unsigned short*)alloc(256 * 384 * 2);
    unsigned short* y1 = (unsigned short*)alloc(256 * 128 * 2);
    unsigned short* y2 = (unsigned short*)alloc(256 * 128 * 2);
    unsigned short* y3 = (unsigned short*)alloc(256 * 128 * 2);
    (void)ws_size;

    auto rpk = [&](const float* W, int Nr, int K, int nT, int kT, unsigned short* o) {
        int tot = nT * kT * 512;
        repack_w_kernel<<<(tot + 255) / 256, 256, 0, stream>>>(W, Nr, K, nT, kT, o);
    };
    rpk(w_ih, 384, 128, 24, 4, wih_pk);
    rpk(w_hh, 384, 128, 24, 4, whh_pk);
    rpk(fc_w, 8192, 128, 512, 4, fcw_pk);
    rpk(fc1_w, 128, 361, 8, 12, f1w_pk);
    rpk(fc2_w, 128, 128, 8, 4, f2w_pk);
    rpk(fc3_w, 8, 128, 1, 4, f3w_pk);

    bn_stats_kernel<<<1024, 256, 0, stream>>>(Xr, Xi, gamma, beta, scale, shift);

    {
        size_t tot = (size_t)256 * 1024 * 128;
        bn_norm_kernel<<<(unsigned)((tot + 255) / 256), 256, 0, stream>>>(Xr, Xi, scale, shift, Xn);
    }

    gru_kernel<<<16, 256, 0, stream>>>(Xn, wih_pk, whh_pk, b_ih, b_hh, hT);

    auto gemm = [&](const unsigned short* A, int lda, const unsigned short* W,
                    const float* bias, int nT, int kT,
                    float* oF, int sF, unsigned short* oB, int sB, int nV, int relu) {
        int waves = 16 * nT;
        gemm_bf16_kernel<<<(waves + 7) / 8, 256, 0, stream>>>(
            A, lda, W, bias, nT, kT, oF, sF, oB, sB, nV, relu);
    };

    // Rx = hT @ fc_w^T + fc_b : (256 x 8192) f32
    gemm(hT, 128, fcw_pk, fc_b, 512, 4, Rx, 8192, nullptr, 0, 8192, 0);

    subspace_kernel<<<256, 256, 0, stream>>>(Rx, Us);

    sv_kernel<<<(361 * 64 + 255) / 256, 256, 0, stream>>>(svr, svi);

    {
        dim3 g(3, 256);
        spectrum_kernel<<<g, 128, 0, stream>>>(Us, svr, svi, spec);
    }

    gemm(spec, 384, f1w_pk, fc1_b, 8, 12, nullptr, 0, y1, 128, 128, 1);
    gemm(y1, 128, f2w_pk, fc2_b, 8, 4, nullptr, 0, y2, 128, 128, 1);
    gemm(y2, 128, f2w_pk, fc2_b, 8, 4, nullptr, 0, y3, 128, 128, 1);

    gemm(y3, 128, f3w_pk, fc3_b, 1, 4, (float*)d_out, 8, nullptr, 0, 8, 0);
}
